// BioRNN_working_60138132079091
// MI455X (gfx1250) — compile-verified, matrix operand order checked by reading, not run on hardware
//
#include <hip/hip_runtime.h>
#include <hip/hip_bf16.h>

// ---------------- problem constants ----------------
#define NN     3840          // total neurons
#define NKK    3968          // K dim of fused matmul = NN + N_IN
#define NIN    128
#define NB     32            // batch
#define NT     100           // timesteps
#define KB     124           // NKK / 32 k-blocks
#define NTN    240           // NN / 16 N-tiles
#define NTM    2             // NB / 16 M-tiles
#define DECAYF 0.2f
#define NOISE_SCALE 0.0063245553f   // sqrt(2*0.2)*0.01
#define EW_BLOCKS 496               // NB*NKK / 256 exactly
#define RO_BLOCKS 12                // 96 readout waves / 8

typedef __bf16 bf16_t;
typedef __attribute__((ext_vector_type(16))) __bf16 v16bf;
typedef __attribute__((ext_vector_type(8)))  float  v8f;

static __device__ __forceinline__ unsigned short f2bf(float f) {
    unsigned u = __float_as_uint(f);
    u += 0x7FFFu + ((u >> 16) & 1u);        // round-to-nearest-even
    return (unsigned short)(u >> 16);
}

// fragment packing: for k-in-block kk (0..31) -> lane-half h, elem e
static __device__ __forceinline__ void kmap(int kk, int& h, int& e) {
    h = (kk >> 3) & 1;
    e = (kk & 7) | ((kk >> 4) << 3);
}

// ---------------- prep: zero state + alpha table ----------------
__global__ __launch_bounds__(256) void k_init(float* hbuf, float* ime, float* alpha) {
    int i = blockIdx.x * 256 + threadIdx.x;
    if (i < NB * NN) hbuf[i] = 0.0f;
    if (i < NB * 1024) ime[i] = 0.0f;
    if (i < 512) {
        // tau = logspace(100..5000, 512); alpha = dt/tau
        float tau = 100.0f * __expf((float)i * (1.0f / 511.0f) * 3.9120230054281460f); // ln(50)
        alpha[i] = 10.0f / tau;
    }
}

// ---------------- prep: pack [w_eff ; w_in] into bf16 WMMA B-tiles ----------------
// B-tile layout: wtile[((j*KB + kb) << 9) + (lane << 4) + e], lane = n%16 + 16*h
__global__ __launch_bounds__(256) void k_pack_w(const float* __restrict__ w_rec,
                                                const float* __restrict__ w_in,
                                                const float* __restrict__ mask,
                                                unsigned short* __restrict__ wtile) {
    int idx = blockIdx.x * 256 + threadIdx.x;
    if (idx >= NKK * NN) return;
    int k = idx / NN;
    int n = idx - k * NN;
    float w;
    if (k < NN) w = fabsf(w_rec[k * NN + n]) * mask[k * NN + n];
    else        w = w_in[(k - NN) * NN + n];
    int j = n >> 4, nn = n & 15;
    int kb = k >> 5, kk = k & 31, h, e;
    kmap(kk, h, e);
    int lane = nn + (h << 4);
    wtile[((j * KB + kb) << 9) + (lane << 4) + e] = f2bf(w);
}

// ---------------- t=0 bootstrap: rates from h0 + A-fragment pack ----------------
__global__ __launch_bounds__(256) void k_rates(const float* __restrict__ hbuf,
                                               const float* __restrict__ x, int t,
                                               float* __restrict__ rbuf,
                                               unsigned short* __restrict__ abuf) {
    int idx = blockIdx.x * 256 + threadIdx.x;
    if (idx >= NB * NKK) return;
    int m = idx / NKK;
    int k = idx - m * NKK;
    float v;
    if (k < NN) {
        float hv = hbuf[m * NN + k];
        bool dend = (k >= 512 && k < 1536) || (k >= 2432 && k < 3456);
        v = dend ? (1.0f / (1.0f + __expf(-hv))) : fmaxf(hv, 0.0f);
        rbuf[m * NN + k] = v;
    } else {
        v = x[(t * NB + m) * NIN + (k - NN)];
    }
    int Mt = m >> 4, mm = m & 15;
    int kb = k >> 5, kk = k & 31, h, e;
    kmap(kk, h, e);
    int lane = mm + (h << 4);
    abuf[((Mt * KB + kb) << 9) + (lane << 4) + e] = f2bf(v);
}

// ---------------- per-step: pre = [r|x] @ [w_eff;w_in] + b via WMMA ----------------
// One wave per N-tile holds BOTH M-tile accumulators: 2 WMMAs per B-fragment load.
__global__ __launch_bounds__(256) void k_matmul(const unsigned short* __restrict__ abuf_,
                                                const unsigned short* __restrict__ wtile_,
                                                const float* __restrict__ bias,
                                                float* __restrict__ pre) {
    const bf16_t* abuf  = (const bf16_t*)abuf_;
    const bf16_t* wtile = (const bf16_t*)wtile_;
    int lane = threadIdx.x & 31;
    int wave = threadIdx.x >> 5;
    int Nt = blockIdx.x * 8 + wave;          // 30 blocks * 8 waves = 240 N-tiles

    const bf16_t* aP0 = abuf + lane * 16;               // M-tile 0
    const bf16_t* aP1 = abuf + (KB << 9) + lane * 16;   // M-tile 1
    const bf16_t* bP  = wtile + ((size_t)(Nt * KB) << 9) + lane * 16;

    v8f acc0 = {0.f, 0.f, 0.f, 0.f, 0.f, 0.f, 0.f, 0.f};
    v8f acc1 = {0.f, 0.f, 0.f, 0.f, 0.f, 0.f, 0.f, 0.f};
#pragma unroll 4
    for (int kb = 0; kb < KB; ++kb) {
        v16bf a0 = *(const v16bf*)(aP0 + (kb << 9));
        v16bf a1 = *(const v16bf*)(aP1 + (kb << 9));
        v16bf bf = *(const v16bf*)(bP  + (kb << 9));
        acc0 = __builtin_amdgcn_wmma_f32_16x16x32_bf16(
            false, a0, false, bf, (short)0, acc0, false, false);
        acc1 = __builtin_amdgcn_wmma_f32_16x16x32_bf16(
            false, a1, false, bf, (short)0, acc1, false, false);
    }
    int n = Nt * 16 + (lane & 15);
    float bb = bias[n];
    int rowBase = (lane >> 4) << 3;
#pragma unroll
    for (int v = 0; v < 8; ++v) {
        pre[(rowBase + v) * NN + n]        = acc0[v] + bb;
        pre[(16 + rowBase + v) * NN + n]   = acc1[v] + bb;
    }
}

// ---------------- per-step fused: readout(t) + update h + rates(t+1) + A-pack ----
// Blocks [0,496): elementwise over NB*NKK; reads r_t (rcur), writes r_{t+1} (rnxt).
// Blocks [496,508): readout from rcur (r_t) — no write conflict with rnxt.
__global__ __launch_bounds__(256) void k_fused(float* __restrict__ hbuf,
                                               float* __restrict__ ime,
                                               const float* __restrict__ pre,
                                               const float* __restrict__ rcur,
                                               float* __restrict__ rnxt,
                                               unsigned short* __restrict__ abuf,
                                               const float* __restrict__ d2s,
                                               const float* __restrict__ alpha,
                                               const float* __restrict__ noise,
                                               const float* __restrict__ x,
                                               const float* __restrict__ w_out,
                                               const float* __restrict__ b_out,
                                               float* __restrict__ out, int t) {
    if (blockIdx.x < EW_BLOCKS) {
        int idx = blockIdx.x * 256 + threadIdx.x;        // exactly NB*NKK threads
        int m = idx / NKK;
        int k = idx - m * NKK;
        float v;
        if (k < NN) {
            int n = k;
            float c = pre[m * NN + n];
            if (n < 512) {                               // SR E-soma
                c += d2s[n]       * rcur[m * NN + 512 + n]
                   + d2s[512 + n] * rcur[m * NN + 1024 + n];
            } else if (n >= 1920 && n < 2432) {          // PFC E-soma
                int j = n - 1920;
                c += d2s[1024 + j] * rcur[m * NN + 2432 + j]
                   + d2s[1536 + j] * rcur[m * NN + 2944 + j];
            } else if (n >= 2432 && n < 3456) {          // PFC E-dend: mGluR
                int j = n - 2432;
                float a = alpha[j & 511];
                float im = ime[m * 1024 + j];
                im = (1.0f - a) * im + a * fmaxf(c, 0.0f);
                ime[m * 1024 + j] = im;
                c += im;
            }
            float hn = (1.0f - DECAYF) * hbuf[m * NN + n] + DECAYF * c
                     + NOISE_SCALE * noise[(size_t)t * NB * NN + m * NN + n];
            hbuf[m * NN + n] = hn;
            bool dend = (n >= 512 && n < 1536) || (n >= 2432 && n < 3456);
            v = dend ? (1.0f / (1.0f + __expf(-hn))) : fmaxf(hn, 0.0f);
            rnxt[m * NN + n] = v;
        } else {
            v = (t + 1 < NT) ? x[((size_t)(t + 1) * NB + m) * NIN + (k - NN)] : 0.0f;
        }
        int Mt = m >> 4, mm = m & 15;
        int kb = k >> 5, kk = k & 31, h, e;
        kmap(kk, h, e);
        int lane = mm + (h << 4);
        abuf[((Mt * KB + kb) << 9) + (lane << 4) + e] = f2bf(v);
    } else {
        // readout: out[t,b,o] = r_t[b,:512] @ w_out[:,o] + b_out[o]
        int lane = threadIdx.x & 31;
        int wid = (blockIdx.x - EW_BLOCKS) * 8 + (threadIdx.x >> 5);
        if (wid >= NB * 3) return;
        int bi = wid / 3, o = wid - bi * 3;
        float s = 0.0f;
#pragma unroll
        for (int k = lane; k < 512; k += 32)
            s += rcur[bi * NN + k] * w_out[k * 3 + o];
#pragma unroll
        for (int off = 16; off > 0; off >>= 1)
            s += __shfl_xor(s, off, 32);
        if (lane == 0) out[t * NB * 3 + bi * 3 + o] = s + b_out[o];
    }
}

// ---------------- host launcher ----------------
extern "C" void kernel_launch(void* const* d_in, const int* in_sizes, int n_in,
                              void* d_out, int out_size, void* d_ws, size_t ws_size,
                              hipStream_t stream) {
    const float* x     = (const float*)d_in[0];   // [T,B,128]
    const float* noise = (const float*)d_in[1];   // [T,B,N]
    const float* w_rec = (const float*)d_in[2];   // [N,N]
    const float* w_in  = (const float*)d_in[3];   // [128,N]
    const float* bias  = (const float*)d_in[4];   // [N]
    const float* d2s   = (const float*)d_in[5];   // [2048]
    const float* w_out = (const float*)d_in[6];   // [512,3]
    const float* b_out = (const float*)d_in[7];   // [3]
    const float* mask  = (const float*)d_in[8];   // [N,N]
    float* out = (float*)d_out;

    // workspace carve-up (256B aligned)
    char* p = (char*)d_ws;
    auto take = [&](size_t bytes) { char* q = p; p += (bytes + 255) & ~(size_t)255; return q; };
    unsigned short* wtile = (unsigned short*)take((size_t)NTN * KB * 512 * 2);  // ~30.5 MB
    unsigned short* abuf  = (unsigned short*)take((size_t)NTM * KB * 512 * 2);
    float* rb0   = (float*)take((size_t)NB * NN * 4);
    float* rb1   = (float*)take((size_t)NB * NN * 4);
    float* pre   = (float*)take((size_t)NB * NN * 4);
    float* hbuf  = (float*)take((size_t)NB * NN * 4);
    float* ime   = (float*)take((size_t)NB * 1024 * 4);
    float* alpha = (float*)take(512 * 4);
    float* rb[2] = {rb0, rb1};

    // one-time prep
    k_init<<<(NB * NN + 255) / 256, 256, 0, stream>>>(hbuf, ime, alpha);
    k_pack_w<<<(NKK * NN + 255) / 256, 256, 0, stream>>>(w_rec, w_in, mask, wtile);
    k_rates<<<EW_BLOCKS, 256, 0, stream>>>(hbuf, x, 0, rb[0], abuf);

    for (int t = 0; t < NT; ++t) {
        k_matmul<<<30, 256, 0, stream>>>(abuf, wtile, bias, pre);
        k_fused<<<EW_BLOCKS + RO_BLOCKS, 256, 0, stream>>>(
            hbuf, ime, pre, rb[t & 1], rb[(t + 1) & 1], abuf,
            d2s, alpha, noise, x, w_out, b_out, out, t);
    }
}
